// GraphTransformerModel_9156870275644
// MI455X (gfx1250) — matrix operand order repaired and checked by reading, not compiled, not run on hardware
//
#include <hip/hip_runtime.h>
#include <hip/hip_bf16.h>

typedef __attribute__((ext_vector_type(4)))  _Float16 v4h;
typedef __attribute__((ext_vector_type(8)))  _Float16 v8h;
typedef __attribute__((ext_vector_type(16))) _Float16 v16h;
typedef __attribute__((ext_vector_type(8)))  float    v8f;

#define NODES 50000
#define EDGES 400000
#define DIN   384
#define HEADS 4
#define CH    128
#define HCC   512   // HEADS*CH

#define WMMA_F16(a, b, c) \
  __builtin_amdgcn_wmma_f32_16x16x32_f16(false, (a), false, (b), (short)0, (c), false, false)

// ---- order-preserving float<->uint encoding for atomic segment-max ----
__device__ __forceinline__ unsigned ord_encode(float f) {
  unsigned u = __float_as_uint(f);
  return (u & 0x80000000u) ? ~u : (u | 0x80000000u);
}
__device__ __forceinline__ float ord_decode(unsigned u) {
  return __uint_as_float((u & 0x80000000u) ? (u & 0x7FFFFFFFu) : ~u);
}
#define ORD_NEG_INF 0x007FFFFFu   // ord_encode(-inf)

// ---------------- elementwise conversion / init kernels ----------------
__global__ void k_f32_to_f16(const float* __restrict__ in, _Float16* __restrict__ out, int n) {
  int i = blockIdx.x * blockDim.x + threadIdx.x;
  if (i < n) out[i] = (_Float16)in[i];
}

__global__ void k_relu_to_f16(const float* __restrict__ in, _Float16* __restrict__ out, int n) {
  int i = blockIdx.x * blockDim.x + threadIdx.x;
  if (i < n) out[i] = (_Float16)fmaxf(in[i], 0.0f);
}

__global__ void k_init_softmax(unsigned* __restrict__ mEnc, float* __restrict__ denom, int n) {
  int i = blockIdx.x * blockDim.x + threadIdx.x;
  if (i < n) { mEnc[i] = ORD_NEG_INF; denom[i] = 0.0f; }
}

// ---------------- WMMA GEMM: out[M,N] = A[M,K] * W[N,K]^T + bias --------
// Register-blocked 16x64 strip per wave32 (1 M-tile x 4 N-tiles): one A
// fragment feeds 4 WMMAs. Software-pipelined with explicit pointer bumps:
// fragments for step i+1 live in fresh registers while step i's WMMAs issue,
// and the 5 address pointers are dedicated registers advanced by +32 elems
// (folds to constant load offsets under unroll) so pointer math never
// aliases fragment registers (avoids WMMA->VALU WAR hazard NOPs).
//
// A fragment (16x32 f16, contiguous k order): lane L, M=L&15, half=L>>4;
//   elems 0..15 = A[M][k0 + half*16 + j]
// B fragment (32x16 f16): lane L, N=L&15, half=L>>4;
//   elems 0..15 = W[N][k0 + half*16 + j]   (K contiguous -> single 32B load)
// D (16x16 f32): acc[r] -> row = r + 8*half, col = L&15
template <bool HALF_OUT>
__global__ void k_gemm_wmma(const _Float16* __restrict__ A,
                            const _Float16* __restrict__ W,
                            const float* __restrict__ bias,
                            void* __restrict__ out,
                            int M, int N, int K) {
  const int stripsN = N >> 6;                       // 64-col super-tiles
  const int totalStrips = (M >> 4) * stripsN;
  const int wave = threadIdx.x >> 5;
  const int lane = threadIdx.x & 31;
  const int strip = blockIdx.x * 8 + wave;
  if (strip >= totalStrips) return;
  const int tm  = strip / stripsN;
  const int tn4 = strip - tm * stripsN;
  const int half = lane >> 4;
  const int lr   = lane & 15;

  const _Float16* __restrict__ ap  = A + (size_t)(tm * 16 + lr) * K + half * 16;
  const size_t wrow = (size_t)(tn4 * 64 + lr) * K + half * 16;
  const _Float16* __restrict__ wp0 = W + wrow;
  const _Float16* __restrict__ wp1 = W + wrow + (size_t)16 * K;
  const _Float16* __restrict__ wp2 = W + wrow + (size_t)32 * K;
  const _Float16* __restrict__ wp3 = W + wrow + (size_t)48 * K;

  union AF { v16h v; v8h h[2]; };

  // prologue: load k-step 0
  AF a;
  a.h[0] = *(const v8h*)(ap);
  a.h[1] = *(const v8h*)(ap + 8);
  v16h b0 = *(const v16h*)(wp0);
  v16h b1 = *(const v16h*)(wp1);
  v16h b2 = *(const v16h*)(wp2);
  v16h b3 = *(const v16h*)(wp3);

  v8f acc0 = {}, acc1 = {}, acc2 = {}, acc3 = {};

  const int kSteps = K >> 5;
#pragma unroll 2
  for (int it = 0; it < kSteps - 1; ++it) {
    ap += 32; wp0 += 32; wp1 += 32; wp2 += 32; wp3 += 32;
    // issue next k-step's loads into fresh registers
    AF an;
    an.h[0] = *(const v8h*)(ap);
    an.h[1] = *(const v8h*)(ap + 8);
    v16h c0 = *(const v16h*)(wp0);
    v16h c1 = *(const v16h*)(wp1);
    v16h c2 = *(const v16h*)(wp2);
    v16h c3 = *(const v16h*)(wp3);
    // compute on current fragments while next loads are in flight
    acc0 = WMMA_F16(a.v, b0, acc0);
    acc1 = WMMA_F16(a.v, b1, acc1);
    acc2 = WMMA_F16(a.v, b2, acc2);
    acc3 = WMMA_F16(a.v, b3, acc3);
    // rotate buffers (register renaming after unroll)
    a.v = an.v; b0 = c0; b1 = c1; b2 = c2; b3 = c3;
  }
  // epilogue: last k-step
  acc0 = WMMA_F16(a.v, b0, acc0);
  acc1 = WMMA_F16(a.v, b1, acc1);
  acc2 = WMMA_F16(a.v, b2, acc2);
  acc3 = WMMA_F16(a.v, b3, acc3);

  auto store_tile = [&](const v8f& acc, int tnLocal) {
    const int col = tn4 * 64 + tnLocal * 16 + lr;
    const float bv = bias[col];
#pragma unroll
    for (int r = 0; r < 8; ++r) {
      const int row = tm * 16 + r + half * 8;
      const float val = acc[r] + bv;
      if (HALF_OUT) ((_Float16*)out)[(size_t)row * N + col] = (_Float16)val;
      else          ((float*)out)[(size_t)row * N + col] = val;
    }
  };
  store_tile(acc0, 0);
  store_tile(acc1, 1);
  store_tile(acc2, 2);
  store_tile(acc3, 3);
}

// ---------------- edge phase ----------------
// alpha[e][h] = dot(q[dst], k[src]) / sqrt(C); segment-max into mEnc[dst][h]
__global__ void k_edge_alpha(const int* __restrict__ src, const int* __restrict__ dst,
                             const _Float16* __restrict__ q, const _Float16* __restrict__ k,
                             float* __restrict__ alpha, unsigned* __restrict__ mEnc, int E) {
  const int gid = blockIdx.x * blockDim.x + threadIdx.x;
  const int e = gid >> 5, lane = gid & 31;
  if (e >= E) return;
  const int s = src[e], d = dst[e];
  const _Float16* __restrict__ qp = q + (size_t)d * HCC;
  const _Float16* __restrict__ kp = k + (size_t)s * HCC;
#pragma unroll
  for (int h = 0; h < HEADS; ++h) {
    v4h qv = *(const v4h*)(qp + h * CH + lane * 4);
    v4h kv = *(const v4h*)(kp + h * CH + lane * 4);
    float sum = 0.0f;
#pragma unroll
    for (int j = 0; j < 4; ++j) sum += (float)qv[j] * (float)kv[j];
#pragma unroll
    for (int off = 16; off >= 1; off >>= 1) sum += __shfl_xor(sum, off, 32);
    if (lane == 0) {
      const float a = sum * 0.08838834764831845f;  // 1/sqrt(128)
      alpha[e * HEADS + h] = a;
      atomicMax(&mEnc[d * HEADS + h], ord_encode(a));
    }
  }
}

// alpha <- exp(alpha - m[dst]); denom[dst] += alpha
__global__ void k_edge_softmax(const int* __restrict__ dst, float* __restrict__ alpha,
                               const unsigned* __restrict__ mEnc, float* __restrict__ denom,
                               int EH) {
  const int i = blockIdx.x * blockDim.x + threadIdx.x;
  if (i >= EH) return;
  const int e = i >> 2, h = i & 3;
  const int d = dst[e];
  const float m = ord_decode(mEnc[d * HEADS + h]);
  const float ev = __expf(alpha[i] - m);
  alpha[i] = ev;
  atomicAdd(&denom[d * HEADS + h], ev);
}

// aggs[dst] += v[src] * (e / denom[dst])   (aggs pre-loaded with skip linear)
__global__ void k_edge_aggregate(const int* __restrict__ src, const int* __restrict__ dst,
                                 const _Float16* __restrict__ v, const float* __restrict__ ealpha,
                                 const float* __restrict__ denom, float* __restrict__ aggs, int E) {
  const int gid = blockIdx.x * blockDim.x + threadIdx.x;
  const int e = gid >> 5, lane = gid & 31;
  if (e >= E) return;
  const int s = src[e], d = dst[e];
#pragma unroll
  for (int h = 0; h < HEADS; ++h) {
    const float a = ealpha[e * HEADS + h] / (denom[d * HEADS + h] + 1e-16f);
    v4h vv = *(const v4h*)(v + (size_t)s * HCC + h * CH + lane * 4);
    float* dp = aggs + (size_t)d * HCC + h * CH + lane * 4;
#pragma unroll
    for (int j = 0; j < 4; ++j) atomicAdd(dp + j, (float)vv[j] * a);
  }
}

// ---------------- host-side launch ----------------
extern "C" void kernel_launch(void* const* d_in, const int* in_sizes, int n_in,
                              void* d_out, int out_size, void* d_ws, size_t ws_size,
                              hipStream_t stream) {
  (void)in_sizes; (void)n_in; (void)out_size; (void)ws_size;
  const float* x    = (const float*)d_in[0];
  const int*   eidx = (const int*)d_in[1];
  const int*   src  = eidx;
  const int*   dst  = eidx + EDGES;

  const float* qw[2] = {(const float*)d_in[2],  (const float*)d_in[10]};
  const float* qb[2] = {(const float*)d_in[3],  (const float*)d_in[11]};
  const float* kw[2] = {(const float*)d_in[4],  (const float*)d_in[12]};
  const float* kb[2] = {(const float*)d_in[5],  (const float*)d_in[13]};
  const float* vw[2] = {(const float*)d_in[6],  (const float*)d_in[14]};
  const float* vb[2] = {(const float*)d_in[7],  (const float*)d_in[15]};
  const float* sw[2] = {(const float*)d_in[8],  (const float*)d_in[16]};
  const float* sb[2] = {(const float*)d_in[9],  (const float*)d_in[17]};
  const float* outw  = (const float*)d_in[18];
  const float* outb  = (const float*)d_in[19];

  // ---- carve workspace (256B-aligned regions) ----
  size_t off = 0;
  auto carve = [&](size_t bytes) -> char* {
    char* p = (char*)d_ws + off;
    off += (bytes + 255) & ~(size_t)255;
    return p;
  };
  _Float16* xh    = (_Float16*)carve((size_t)NODES * HCC * 2);  // current-layer input (f16)
  _Float16* qh    = (_Float16*)carve((size_t)NODES * HCC * 2);
  _Float16* kh    = (_Float16*)carve((size_t)NODES * HCC * 2);
  _Float16* vh    = (_Float16*)carve((size_t)NODES * HCC * 2);
  float*    aggs  = (float*)   carve((size_t)NODES * HCC * 4);  // skip + messages (f32)
  float*    alpha = (float*)   carve((size_t)EDGES * HEADS * 4);
  unsigned* mEnc  = (unsigned*)carve((size_t)NODES * HEADS * 4);
  float*    denom = (float*)   carve((size_t)NODES * HEADS * 4);
  _Float16* qwh[2], *kwh[2], *vwh[2], *swh[2];
  const int wElems[2] = {HCC * DIN, HCC * HCC};
  for (int l = 0; l < 2; ++l) {
    qwh[l] = (_Float16*)carve((size_t)wElems[l] * 2);
    kwh[l] = (_Float16*)carve((size_t)wElems[l] * 2);
    vwh[l] = (_Float16*)carve((size_t)wElems[l] * 2);
    swh[l] = (_Float16*)carve((size_t)wElems[l] * 2);
  }
  _Float16* outwh = (_Float16*)carve((size_t)DIN * HCC * 2);

  auto cvt = [&](const float* in, _Float16* out, int n) {
    k_f32_to_f16<<<(n + 255) / 256, 256, 0, stream>>>(in, out, n);
  };
  auto gemm = [&](const _Float16* A, const _Float16* W, const float* bias, void* out,
                  int M, int N, int K, bool halfOut) {
    const int blocks = ((M / 16) * (N / 64) + 7) / 8;   // 16x64 strip per wave, 8 waves/block
    if (halfOut) k_gemm_wmma<true ><<<blocks, 256, 0, stream>>>(A, W, bias, out, M, N, K);
    else         k_gemm_wmma<false><<<blocks, 256, 0, stream>>>(A, W, bias, out, M, N, K);
  };

  // ---- one-time conversions ----
  cvt(x, xh, NODES * DIN);
  for (int l = 0; l < 2; ++l) {
    cvt(qw[l], qwh[l], wElems[l]);
    cvt(kw[l], kwh[l], wElems[l]);
    cvt(vw[l], vwh[l], wElems[l]);
    cvt(sw[l], swh[l], wElems[l]);
  }
  cvt(outw, outwh, DIN * HCC);

  // ---- two TransformerConv layers ----
  for (int l = 0; l < 2; ++l) {
    const int K = (l == 0) ? DIN : HCC;
    gemm(xh, qwh[l], qb[l], qh,   NODES, HCC, K, true);
    gemm(xh, kwh[l], kb[l], kh,   NODES, HCC, K, true);
    gemm(xh, vwh[l], vb[l], vh,   NODES, HCC, K, true);
    gemm(xh, swh[l], sb[l], aggs, NODES, HCC, K, false);  // skip linear seeds aggs

    k_init_softmax<<<(NODES * HEADS + 255) / 256, 256, 0, stream>>>(mEnc, denom, NODES * HEADS);
    k_edge_alpha    <<<(EDGES * 32 + 255) / 256, 256, 0, stream>>>(src, dst, qh, kh, alpha, mEnc, EDGES);
    k_edge_softmax  <<<(EDGES * HEADS + 255) / 256, 256, 0, stream>>>(dst, alpha, mEnc, denom, EDGES * HEADS);
    k_edge_aggregate<<<(EDGES * 32 + 255) / 256, 256, 0, stream>>>(src, dst, vh, alpha, denom, aggs, EDGES);

    // relu + downconvert becomes next layer's input
    k_relu_to_f16<<<(NODES * HCC + 255) / 256, 256, 0, stream>>>(aggs, xh, NODES * HCC);
  }

  // ---- output projection (f32 result) ----
  gemm(xh, outwh, outb, d_out, NODES, DIN, HCC, false);
}
